// S2S_1821066134091
// MI455X (gfx1250) — compile-verified, matrix operand order checked by reading, not run on hardware
//
#include <hip/hip_runtime.h>

// ---------------------------------------------------------------------------
// Problem constants (match the reference)
// ---------------------------------------------------------------------------
#define S_  128
#define B_  64
#define T_  128
#define E_  256
#define H_  512
#define V_  10000
#define G4_ 2048   // 4*H

typedef __bf16 bf16_t;
typedef __attribute__((ext_vector_type(16))) __bf16 v16bf;
typedef __attribute__((ext_vector_type(8)))  float  v8f;

__device__ inline bf16_t f2bf(float x) {
    unsigned u = __float_as_uint(x);
    unsigned r = (u + 0x7FFFu + ((u >> 16) & 1u)) >> 16;   // round-to-nearest-even
    unsigned short s = (unsigned short)r;
    return __builtin_bit_cast(bf16_t, s);
}

// Load one 16x32 bf16 fragment (A-layout / B-as-weight-rows layout).
// p = base + (tile_row + (lane&15)) * ld + ((lane>>4)<<3)
__device__ inline v16bf load_frag(const bf16_t* __restrict__ p) {
    v16bf v;
#pragma unroll
    for (int i = 0; i < 8; ++i) { v[i] = p[i]; v[i + 8] = p[16 + i]; }
    return v;
}

// ---------------------------------------------------------------------------
// Dual-phase WMMA GEMM (one 16x16 C tile per wave):
//   C = A1(MxK1) * B1(NxK1)^T + A2(MxK2) * B2(NxK2)^T + bias
// Used for the fused LSTM-step GEMMs (x*Wih^T + h*Whh^T) and, with K2=0,
// as a plain single GEMM. All matrices bf16 row-major; C fp32.
// ---------------------------------------------------------------------------
__global__ void wmma_step_gemm(const bf16_t* __restrict__ A1, int lda1,
                               const bf16_t* __restrict__ B1, int ldb1, int K1,
                               const bf16_t* __restrict__ A2, int lda2,
                               const bf16_t* __restrict__ B2, int ldb2, int K2,
                               const float* __restrict__ bias,
                               float* __restrict__ C, int ldc, int M, int N)
{
    int wave = blockIdx.x * (blockDim.x >> 5) + (threadIdx.x >> 5);
    int lane = threadIdx.x & 31;
    int tiles_n = N >> 4;
    int total   = (M >> 4) * tiles_n;
    if (wave >= total) return;                  // wave-uniform: EXEC stays all-ones

    int tm = wave / tiles_n, tn = wave % tiles_n;
    int m0 = tm << 4, n0 = tn << 4;
    int r  = lane & 15;
    int kh = (lane >> 4) << 3;
    int row_off = (lane >> 4) << 3;
    int col = lane & 15;

    v8f acc;
#pragma unroll
    for (int i = 0; i < 8; ++i) acc[i] = 0.f;

    const bf16_t* pa = A1 + (size_t)(m0 + r) * lda1 + kh;
    const bf16_t* pb = B1 + (size_t)(n0 + r) * ldb1 + kh;
    for (int k = 0; k < K1; k += 32) {
        acc = __builtin_amdgcn_wmma_f32_16x16x32_bf16(false, load_frag(pa),
                                                      false, load_frag(pb),
                                                      (short)0, acc, false, false);
        pa += 32; pb += 32;
    }
    if (K2 > 0) {
        pa = A2 + (size_t)(m0 + r) * lda2 + kh;
        pb = B2 + (size_t)(n0 + r) * ldb2 + kh;
        for (int k = 0; k < K2; k += 32) {
            acc = __builtin_amdgcn_wmma_f32_16x16x32_bf16(false, load_frag(pa),
                                                          false, load_frag(pb),
                                                          (short)0, acc, false, false);
            pa += 32; pb += 32;
        }
    }

    float bc = bias ? bias[n0 + col] : 0.f;
#pragma unroll
    for (int i = 0; i < 8; ++i)
        C[(size_t)(m0 + i + row_off) * ldc + n0 + col] = acc[i] + bc;
}

// ---------------------------------------------------------------------------
// Register-blocked WMMA GEMM: each wave computes a 64x32 C block
// (4 M-tiles x 2 N-tiles -> 8 WMMAs per 6 fragment loads per K-step).
// M must be a multiple of 64. N tail (non-multiple-of-32) handled by a
// wave-uniform single-tile path. Used for the big vocab projection.
// ---------------------------------------------------------------------------
__global__ void wmma_gemm_blk(const bf16_t* __restrict__ A, int lda,
                              const bf16_t* __restrict__ Bw, int ldb,
                              const float* __restrict__ bias,
                              float* __restrict__ C, int ldc,
                              int M, int N, int K)
{
    int wave = blockIdx.x * (blockDim.x >> 5) + (threadIdx.x >> 5);
    int lane = threadIdx.x & 31;
    int nblk = (N + 31) >> 5;
    int total = (M >> 6) * nblk;
    if (wave >= total) return;                  // wave-uniform

    int mb = wave / nblk, nb = wave % nblk;
    int m0 = mb << 6, n0 = nb << 5;
    bool two = (n0 + 32 <= N);                  // wave-uniform predicate

    int r  = lane & 15;
    int kh = (lane >> 4) << 3;
    int row_off = (lane >> 4) << 3;
    int col = lane & 15;

    v8f acc[8];
#pragma unroll
    for (int t = 0; t < 8; ++t)
#pragma unroll
        for (int i = 0; i < 8; ++i) acc[t][i] = 0.f;

    const bf16_t* pa0 = A + (size_t)(m0 + r) * lda + kh;
    const bf16_t* pb0 = Bw + (size_t)(n0 + r) * ldb + kh;

    for (int k = 0; k < K; k += 32) {
        v16bf af[4];
#pragma unroll
        for (int mi = 0; mi < 4; ++mi)
            af[mi] = load_frag(pa0 + (size_t)(mi * 16) * lda + k);
        v16bf b0 = load_frag(pb0 + k);
#pragma unroll
        for (int mi = 0; mi < 4; ++mi)
            acc[mi * 2] = __builtin_amdgcn_wmma_f32_16x16x32_bf16(
                false, af[mi], false, b0, (short)0, acc[mi * 2], false, false);
        if (two) {
            v16bf b1 = load_frag(pb0 + (size_t)16 * ldb + k);
#pragma unroll
            for (int mi = 0; mi < 4; ++mi)
                acc[mi * 2 + 1] = __builtin_amdgcn_wmma_f32_16x16x32_bf16(
                    false, af[mi], false, b1, (short)0, acc[mi * 2 + 1], false, false);
        }
    }

    float bc0 = bias ? bias[n0 + col] : 0.f;
    float bc1 = (two && bias) ? bias[n0 + 16 + col] : 0.f;
#pragma unroll
    for (int mi = 0; mi < 4; ++mi) {
#pragma unroll
        for (int i = 0; i < 8; ++i) {
            size_t row = (size_t)(m0 + mi * 16 + i + row_off);
            C[row * ldc + n0 + col] = acc[mi * 2][i] + bc0;
        }
        if (two) {
#pragma unroll
            for (int i = 0; i < 8; ++i) {
                size_t row = (size_t)(m0 + mi * 16 + i + row_off);
                C[row * ldc + n0 + 16 + col] = acc[mi * 2 + 1][i] + bc1;
            }
        }
    }
}

// ---------------------------------------------------------------------------
// Elementwise LSTM gate update. g: (B_ x 4H) pre-activations [i|f|g|o].
// Updates c (in place), h (fp32), h_bf (bf16); optional extra fp32/bf16 sinks.
// ---------------------------------------------------------------------------
__global__ void lstm_update(const float* __restrict__ g, float* __restrict__ c,
                            float* __restrict__ h, bf16_t* __restrict__ h_bf,
                            float* aux_f, int aux_stride,
                            bf16_t* aux_bf, int auxbf_stride)
{
    int idx = blockIdx.x * blockDim.x + threadIdx.x;
    if (idx >= B_ * H_) return;
    int b = idx >> 9, j = idx & (H_ - 1);
    const float* gr = g + (size_t)b * G4_;
    float gi = gr[j], gf = gr[H_ + j], gc = gr[2 * H_ + j], go = gr[3 * H_ + j];
    float si = 1.f / (1.f + expf(-gi));
    float sf = 1.f / (1.f + expf(-gf));
    float so = 1.f / (1.f + expf(-go));
    float c2 = sf * c[idx] + si * tanhf(gc);
    float h2 = so * tanhf(c2);
    c[idx] = c2;
    h[idx] = h2;
    h_bf[idx] = f2bf(h2);
    if (aux_f)  aux_f [(size_t)b * aux_stride   + j] = h2;
    if (aux_bf) aux_bf[(size_t)b * auxbf_stride + j] = f2bf(h2);
}

// ---------------------------------------------------------------------------
// enc_score[s,b] = dot(enc_states[s,b,:2H], w_enc) + energy_b   (one wave/row)
// ---------------------------------------------------------------------------
__global__ void enc_score_kernel(const float* __restrict__ enc_states,
                                 const float* __restrict__ energy_W,
                                 const float* __restrict__ energy_b,
                                 float* __restrict__ enc_score)
{
    int row  = blockIdx.x * (blockDim.x >> 5) + (threadIdx.x >> 5);
    int lane = threadIdx.x & 31;
    if (row >= S_ * B_) return;
    const float* p = enc_states + (size_t)row * (2 * H_);
    const float* w = energy_W + H_;          // w_enc
    float s = 0.f;
    for (int k = lane; k < 2 * H_; k += 32) s += p[k] * w[k];
#pragma unroll
    for (int off = 16; off > 0; off >>= 1) s += __shfl_down(s, off, 32);
    if (lane == 0) enc_score[row] = s + energy_b[0];
}

// ---------------------------------------------------------------------------
// Attention + context + rnn_in assembly for one decoder step (one block per b).
// ---------------------------------------------------------------------------
__global__ void attention_kernel(const float* __restrict__ h_dec,
                                 const float* __restrict__ enc_score,
                                 const float* __restrict__ enc_states,
                                 const float* __restrict__ energy_W,   // w_h = [:H]
                                 const bf16_t* __restrict__ xe_bf,
                                 int t, bf16_t* __restrict__ rnn_in_bf)
{
    int b = blockIdx.x, tid = threadIdx.x;   // 128 threads, S_ == 128
    __shared__ float sh[128];
    __shared__ float attn[128];

    float part = 0.f;
    for (int k = tid; k < H_; k += 128) part += h_dec[b * H_ + k] * energy_W[k];
    sh[tid] = part; __syncthreads();
    for (int s = 64; s > 0; s >>= 1) { if (tid < s) sh[tid] += sh[tid + s]; __syncthreads(); }
    float hdot = sh[0]; __syncthreads();

    float e = fmaxf(hdot + enc_score[tid * B_ + b], 0.f);     // relu, s = tid
    sh[tid] = e; __syncthreads();
    for (int s = 64; s > 0; s >>= 1) { if (tid < s) sh[tid] = fmaxf(sh[tid], sh[tid + s]); __syncthreads(); }
    float mx = sh[0]; __syncthreads();
    float ex = expf(e - mx);
    sh[tid] = ex; __syncthreads();
    for (int s = 64; s > 0; s >>= 1) { if (tid < s) sh[tid] += sh[tid + s]; __syncthreads(); }
    float denom = sh[0]; __syncthreads();
    attn[tid] = ex / denom; __syncthreads();

    // ctx[b,d] = sum_s attn[s] * enc_states[s,b,d], d in [0, 2H)
    for (int d = tid; d < 2 * H_; d += 128) {
        float acc = 0.f;
        for (int s = 0; s < S_; ++s)
            acc += attn[s] * enc_states[((size_t)s * B_ + b) * (2 * H_) + d];
        rnn_in_bf[(size_t)b * (2 * H_ + E_) + d] = f2bf(acc);
    }
    for (int e2 = tid; e2 < E_; e2 += 128)
        rnn_in_bf[(size_t)b * (2 * H_ + E_) + 2 * H_ + e2] =
            xe_bf[((size_t)t * B_ + b) * E_ + e2];
}

// ---------------------------------------------------------------------------
// Small utility kernels
// ---------------------------------------------------------------------------
__global__ void zero_u32(unsigned* __restrict__ p, long n) {
    long i = (long)blockIdx.x * blockDim.x + threadIdx.x;
    if (i < n) p[i] = 0u;
}
__global__ void cvt_f32_bf16(const float* __restrict__ in, bf16_t* __restrict__ out, long n) {
    long i = (long)blockIdx.x * blockDim.x + threadIdx.x;
    if (i < n) out[i] = f2bf(in[i]);
}
__global__ void add2(const float* __restrict__ a, const float* __restrict__ b,
                     float* __restrict__ out, int n) {
    int i = blockIdx.x * blockDim.x + threadIdx.x;
    if (i < n) out[i] = a[i] + b[i];
}
__global__ void gather_rows_bf16(const float* __restrict__ table, const int* __restrict__ tok,
                                 bf16_t* __restrict__ out, long total, int ecols) {
    long idx = (long)blockIdx.x * blockDim.x + threadIdx.x;
    if (idx >= total) return;
    long row = idx / ecols; int e = (int)(idx % ecols);
    out[idx] = f2bf(table[(size_t)tok[row] * ecols + e]);
}
__global__ void concat2_bf16(const float* __restrict__ a, const float* __restrict__ b,
                             bf16_t* __restrict__ out) {
    int idx = blockIdx.x * blockDim.x + threadIdx.x;   // B_ * 2H
    if (idx >= B_ * 2 * H_) return;
    int row = idx >> 10, j = idx & (2 * H_ - 1);
    float v = (j < H_) ? a[row * H_ + j] : b[row * H_ + j - H_];
    out[idx] = f2bf(v);
}

// ---------------------------------------------------------------------------
// Host side
// ---------------------------------------------------------------------------
static inline size_t align_up(size_t x) { return (x + 255) & ~(size_t)255; }
#define DIV_UP(a, b) (((a) + (b) - 1) / (b))

extern "C" void kernel_launch(void* const* d_in, const int* in_sizes, int n_in,
                              void* d_out, int out_size, void* d_ws, size_t ws_size,
                              hipStream_t stream) {
    (void)in_sizes; (void)n_in; (void)out_size; (void)ws_size;

    const int*   sursa      = (const int*)  d_in[0];
    const int*   target     = (const int*)  d_in[1];
    const float* enc_emb    = (const float*)d_in[2];
    const float* enc_Wih_f  = (const float*)d_in[3];
    const float* enc_Whh_f  = (const float*)d_in[4];
    const float* enc_bih_f  = (const float*)d_in[5];
    const float* enc_bhh_f  = (const float*)d_in[6];
    const float* enc_Wih_b  = (const float*)d_in[7];
    const float* enc_Whh_b  = (const float*)d_in[8];
    const float* enc_bih_b  = (const float*)d_in[9];
    const float* enc_bhh_b  = (const float*)d_in[10];
    const float* fc_hidden_W= (const float*)d_in[11];
    const float* fc_hidden_b= (const float*)d_in[12];
    const float* fc_cell_W  = (const float*)d_in[13];
    const float* fc_cell_b  = (const float*)d_in[14];
    const float* dec_emb    = (const float*)d_in[15];
    const float* dec_Wih    = (const float*)d_in[16];
    const float* dec_Whh    = (const float*)d_in[17];
    const float* dec_bih    = (const float*)d_in[18];
    const float* dec_bhh    = (const float*)d_in[19];
    const float* energy_W   = (const float*)d_in[20];
    const float* energy_b   = (const float*)d_in[21];
    const float* fc_W       = (const float*)d_in[22];
    const float* fc_b       = (const float*)d_in[23];

    float* out = (float*)d_out;

    // --- workspace carve-up ---
    char* ws = (char*)d_ws; size_t off = 0;
    auto alloc = [&](size_t bytes) -> void* {
        void* p = ws + off; off = align_up(off + bytes); return p;
    };
    const int KDEC = 2 * H_ + E_;  // 1280

    bf16_t* emb_bf   = (bf16_t*)alloc((size_t)S_ * B_ * E_ * 2);
    bf16_t* xe_bf    = (bf16_t*)alloc((size_t)(T_ - 1) * B_ * E_ * 2);
    bf16_t* Wih_f_bf = (bf16_t*)alloc((size_t)G4_ * E_ * 2);
    bf16_t* Whh_f_bf = (bf16_t*)alloc((size_t)G4_ * H_ * 2);
    bf16_t* Wih_b_bf = (bf16_t*)alloc((size_t)G4_ * E_ * 2);
    bf16_t* Whh_b_bf = (bf16_t*)alloc((size_t)G4_ * H_ * 2);
    bf16_t* dWih_bf  = (bf16_t*)alloc((size_t)G4_ * KDEC * 2);
    bf16_t* dWhh_bf  = (bf16_t*)alloc((size_t)G4_ * H_ * 2);
    bf16_t* fchW_bf  = (bf16_t*)alloc((size_t)H_ * 2 * H_ * 2);
    bf16_t* fccW_bf  = (bf16_t*)alloc((size_t)H_ * 2 * H_ * 2);
    bf16_t* fcW_bf   = (bf16_t*)alloc((size_t)V_ * H_ * 2);
    bf16_t* h_f_bf   = (bf16_t*)alloc((size_t)B_ * H_ * 2);
    bf16_t* h_b_bf   = (bf16_t*)alloc((size_t)B_ * H_ * 2);
    bf16_t* h_dec_bf = (bf16_t*)alloc((size_t)B_ * H_ * 2);
    bf16_t* hcat_bf  = (bf16_t*)alloc((size_t)B_ * 2 * H_ * 2);
    bf16_t* ccat_bf  = (bf16_t*)alloc((size_t)B_ * 2 * H_ * 2);
    bf16_t* rnn_in_bf= (bf16_t*)alloc((size_t)B_ * KDEC * 2);
    bf16_t* Hs_dec_bf= (bf16_t*)alloc((size_t)(T_ - 1) * B_ * H_ * 2);
    float* bias_f    = (float*)alloc(G4_ * 4);
    float* bias_b    = (float*)alloc(G4_ * 4);
    float* bias_dec  = (float*)alloc(G4_ * 4);
    float* g         = (float*)alloc((size_t)B_ * G4_ * 4);
    float* h_f       = (float*)alloc((size_t)B_ * H_ * 4);
    float* c_f       = (float*)alloc((size_t)B_ * H_ * 4);
    float* h_b       = (float*)alloc((size_t)B_ * H_ * 4);
    float* c_b       = (float*)alloc((size_t)B_ * H_ * 4);
    float* h_dec     = (float*)alloc((size_t)B_ * H_ * 4);
    float* c_dec     = (float*)alloc((size_t)B_ * H_ * 4);
    float* enc_states= (float*)alloc((size_t)S_ * B_ * 2 * H_ * 4);
    float* enc_sc    = (float*)alloc((size_t)S_ * B_ * 4);

    // --- init: zero out[0] timestep + LSTM states ---
    {
        long n = (long)B_ * V_;
        zero_u32<<<DIV_UP(n, 256), 256, 0, stream>>>((unsigned*)out, n);
        long ns = (long)B_ * H_;
        zero_u32<<<DIV_UP(ns, 256), 256, 0, stream>>>((unsigned*)h_f, ns);
        zero_u32<<<DIV_UP(ns, 256), 256, 0, stream>>>((unsigned*)c_f, ns);
        zero_u32<<<DIV_UP(ns, 256), 256, 0, stream>>>((unsigned*)h_b, ns);
        zero_u32<<<DIV_UP(ns, 256), 256, 0, stream>>>((unsigned*)c_b, ns);
        long nh = (long)B_ * H_ / 2;  // bf16 buffers as u32 words
        zero_u32<<<DIV_UP(nh, 256), 256, 0, stream>>>((unsigned*)h_f_bf, nh);
        zero_u32<<<DIV_UP(nh, 256), 256, 0, stream>>>((unsigned*)h_b_bf, nh);
    }

    // --- weight conversion to bf16 ---
    auto cvt = [&](const float* src, bf16_t* dst, long n) {
        cvt_f32_bf16<<<DIV_UP(n, 256), 256, 0, stream>>>(src, dst, n);
    };
    cvt(enc_Wih_f,  Wih_f_bf, (long)G4_ * E_);
    cvt(enc_Whh_f,  Whh_f_bf, (long)G4_ * H_);
    cvt(enc_Wih_b,  Wih_b_bf, (long)G4_ * E_);
    cvt(enc_Whh_b,  Whh_b_bf, (long)G4_ * H_);
    cvt(dec_Wih,    dWih_bf,  (long)G4_ * KDEC);
    cvt(dec_Whh,    dWhh_bf,  (long)G4_ * H_);
    cvt(fc_hidden_W,fchW_bf,  (long)H_ * 2 * H_);
    cvt(fc_cell_W,  fccW_bf,  (long)H_ * 2 * H_);
    cvt(fc_W,       fcW_bf,   (long)V_ * H_);

    add2<<<DIV_UP(G4_, 256), 256, 0, stream>>>(enc_bih_f, enc_bhh_f, bias_f, G4_);
    add2<<<DIV_UP(G4_, 256), 256, 0, stream>>>(enc_bih_b, enc_bhh_b, bias_b, G4_);
    add2<<<DIV_UP(G4_, 256), 256, 0, stream>>>(dec_bih,   dec_bhh,   bias_dec, G4_);

    // --- embedding gathers (fp32 table -> bf16 rows) ---
    {
        long n1 = (long)S_ * B_ * E_;
        gather_rows_bf16<<<DIV_UP(n1, 256), 256, 0, stream>>>(enc_emb, sursa, emb_bf, n1, E_);
        long n2 = (long)(T_ - 1) * B_ * E_;
        gather_rows_bf16<<<DIV_UP(n2, 256), 256, 0, stream>>>(dec_emb, target, xe_bf, n2, E_);
    }

    // --- bidirectional encoder: fused (x*Wih^T + h*Whh^T + bias) per step ---
    const int step_blk = DIV_UP((B_ / 16) * (G4_ / 16), 8);   // 512 tiles -> 64 blocks
    const int upd_blk  = DIV_UP(B_ * H_, 256);
    for (int t = 0; t < S_; ++t) {
        // forward dir
        wmma_step_gemm<<<step_blk, 256, 0, stream>>>(
            emb_bf + (size_t)t * B_ * E_, E_, Wih_f_bf, E_, E_,
            h_f_bf, H_, Whh_f_bf, H_, H_,
            bias_f, g, G4_, B_, G4_);
        lstm_update<<<upd_blk, 256, 0, stream>>>(
            g, c_f, h_f, h_f_bf, enc_states + (size_t)t * B_ * 2 * H_, 2 * H_, nullptr, 0);
        // backward dir (consumes emb[S-1-t], writes enc_states slice [S-1-t, :, H:2H])
        int tb = S_ - 1 - t;
        wmma_step_gemm<<<step_blk, 256, 0, stream>>>(
            emb_bf + (size_t)tb * B_ * E_, E_, Wih_b_bf, E_, E_,
            h_b_bf, H_, Whh_b_bf, H_, H_,
            bias_b, g, G4_, B_, G4_);
        lstm_update<<<upd_blk, 256, 0, stream>>>(
            g, c_b, h_b, h_b_bf, enc_states + (size_t)tb * B_ * 2 * H_ + H_, 2 * H_, nullptr, 0);
    }

    // --- initial decoder state: h_dec = [hf|hb] @ fc_hidden_W^T + b, same for c ---
    concat2_bf16<<<DIV_UP(B_ * 2 * H_, 256), 256, 0, stream>>>(h_f, h_b, hcat_bf);
    concat2_bf16<<<DIV_UP(B_ * 2 * H_, 256), 256, 0, stream>>>(c_f, c_b, ccat_bf);
    {
        int blk = DIV_UP((B_ / 16) * (H_ / 16), 8);   // 128 tiles -> 16 blocks
        wmma_step_gemm<<<blk, 256, 0, stream>>>(
            hcat_bf, 2 * H_, fchW_bf, 2 * H_, 2 * H_,
            nullptr, 0, nullptr, 0, 0,
            fc_hidden_b, h_dec, H_, B_, H_);
        wmma_step_gemm<<<blk, 256, 0, stream>>>(
            ccat_bf, 2 * H_, fccW_bf, 2 * H_, 2 * H_,
            nullptr, 0, nullptr, 0, 0,
            fc_cell_b, c_dec, H_, B_, H_);
    }
    cvt_f32_bf16<<<DIV_UP((long)B_ * H_, 256), 256, 0, stream>>>(h_dec, h_dec_bf, (long)B_ * H_);

    // --- precompute enc_score = enc_states . w_enc + energy_b ---
    enc_score_kernel<<<DIV_UP(S_ * B_, 8), 256, 0, stream>>>(enc_states, energy_W, energy_b, enc_sc);

    // --- attention decoder: fused (rnn_in*dWih^T + h*dWhh^T + bias) per step ---
    for (int t = 0; t < T_ - 1; ++t) {
        attention_kernel<<<B_, 128, 0, stream>>>(
            h_dec, enc_sc, enc_states, energy_W, xe_bf, t, rnn_in_bf);
        wmma_step_gemm<<<step_blk, 256, 0, stream>>>(
            rnn_in_bf, KDEC, dWih_bf, KDEC, KDEC,
            h_dec_bf, H_, dWhh_bf, H_, H_,
            bias_dec, g, G4_, B_, G4_);
        lstm_update<<<upd_blk, 256, 0, stream>>>(
            g, c_dec, h_dec, h_dec_bf, nullptr, 0, Hs_dec_bf + (size_t)t * B_ * H_, H_);
    }

    // --- batched vocab projection (register-blocked 64x32 per wave):
    //     preds = Hs_dec @ fc_W^T + fc_b -> out[1:, :, :]
    {
        int M = (T_ - 1) * B_;                             // 8128 = 127 * 64
        int nblk = DIV_UP(V_, 32);                         // 313 (last block: 1 tile)
        int waves = (M / 64) * nblk;                       // 39751
        wmma_gemm_blk<<<DIV_UP(waves, 8), 256, 0, stream>>>(
            Hs_dec_bf, H_, fcW_bf, H_, fc_b, out + (size_t)B_ * V_, V_, M, V_, H_);
    }
}
